// DynamicEdgeWeightLearner_29317446762899
// MI455X (gfx1250) — matrix operand order repaired and checked by reading, not compiled, not run on hardware
//
#include <hip/hip_runtime.h>

typedef __attribute__((ext_vector_type(16))) _Float16 v16h;
typedef __attribute__((ext_vector_type(8)))  _Float16 v8h;
typedef __attribute__((ext_vector_type(4)))  _Float16 v4h;
typedef __attribute__((ext_vector_type(2)))  _Float16 v2h;
typedef __attribute__((ext_vector_type(8)))  float    v8f;
typedef __attribute__((ext_vector_type(4)))  float    v4f;

#define WAVES   8
#define A_ROW   136   // 128 + 8 f16 pad -> 272B rows, conflict-free b128 frag loads
#define W1_ROW  136
#define H1_ROW  40    // 32 + 8 f16 pad -> 80B rows
#define W2_ROW  40

static __device__ __forceinline__ v16h cat8(v8h a, v8h b) {
  return __builtin_shufflevector(a, b, 0,1,2,3,4,5,6,7,8,9,10,11,12,13,14,15);
}
static __device__ __forceinline__ v8f splat8(float s) {
  v8f r = {s,s,s,s,s,s,s,s};
  return r;
}
// single-instruction ReLU: v_med3_f32(x, 0, huge)
static __device__ __forceinline__ float relu(float v) {
  return __builtin_amdgcn_fmed3f(v, 0.0f, 3.0e38f);
}
// fast sigmoid: v_exp_f32 + v_rcp_f32, no precise-div chain
static __device__ __forceinline__ float sigm(float v) {
  return __builtin_amdgcn_rcpf(1.0f + __expf(-v));
}

__global__ __launch_bounds__(256)
void edge_mlp_wmma_kernel(const float* __restrict__ x,
                          const int*   __restrict__ ei,     // [2][E] int32
                          const float* __restrict__ W1,     // [32][128]
                          const float* __restrict__ b1,     // [32]
                          const float* __restrict__ W2,     // [16][32]
                          const float* __restrict__ b2,     // [16]
                          const float* __restrict__ W3,     // [1][16]
                          const float* __restrict__ b3,     // [1]
                          const float* __restrict__ temp,   // [1]
                          float* __restrict__ out,          // [E]
                          int E, int nTiles)
{
  __shared__ _Float16 sW1[32 * W1_ROW];          // 8704 B
  __shared__ _Float16 sW2[16 * W2_ROW];          // 1280 B
  __shared__ _Float16 sA [WAVES * 16 * A_ROW];   // 34816 B  (per-wave 16x128 A tiles)
  __shared__ _Float16 sH1[WAVES * 16 * H1_ROW];  // 10240 B  (per-wave h1/h2 restage)

  const int tid  = threadIdx.x;
  const int w    = tid >> 5;      // wave id (wave32)
  const int lane = tid & 31;
  const int lo   = lane & 15;
  const int hi   = lane >> 4;     // lane half

  // --- stage weights f32 -> f16 into padded LDS (once per block) ---
  for (int i = tid; i < 32 * 128; i += 256)
    sW1[(i >> 7) * W1_ROW + (i & 127)] = (_Float16)W1[i];
  for (int i = tid; i < 16 * 32; i += 256)
    sW2[(i >> 5) * W2_ROW + (i & 31)] = (_Float16)W2[i];
  __syncthreads();

  // --- W1 B fragments (contiguous K order, same as A: lane = N column,
  //     lane-half selects K 0-15 vs 16-31, K sequential within the halves) ---
  v16h bw1[2][4];
  #pragma unroll
  for (int nt = 0; nt < 2; ++nt) {
    #pragma unroll
    for (int kk = 0; kk < 4; ++kk) {
      const _Float16* p = &sW1[(nt * 16 + lo) * W1_ROW + kk * 32 + hi * 16];
      bw1[nt][kk] = cat8(*(const v8h*)p, *(const v8h*)(p + 8));
    }
  }

  // --- W2 B fragment with K-permutation pi(s) = (s&1)*16 + (s>>1).
  //     h1 is stored interleaved (pairs (k, k+16) adjacent) so the C->A
  //     restage packs into b32 stores; applying the same pi here keeps
  //     A-slot/B-slot features matched, so the GEMM is unchanged. ---
  v16h bw2;
  #pragma unroll
  for (int t = 0; t < 16; ++t) {
    const int hwk  = hi * 16 + t;
    const int feat = ((hwk & 1) << 4) | (hwk >> 1);
    bw2[t] = sW2[lo * W2_ROW + feat];
  }

  // GEMM3 B fragment: W3 in column n=0, K=0..15 (lane 0 only), zero elsewhere.
  v16h bw3 = {};
  if (lane == 0) {
    #pragma unroll
    for (int j = 0; j < 16; ++j) bw3[j] = (_Float16)W3[j];
  }

  const float bias1a = b1[lo];
  const float bias1b = b1[16 + lo];
  const float bias2  = b2[lo];
  const float b3v    = b3[0];
  const float itemp  = __builtin_amdgcn_rcpf(temp[0]);

  _Float16* aRow  = &sA [w * 16 * A_ROW];
  _Float16* h1Row = &sH1[w * 16 * H1_ROW];
  const char* xb  = (const char*)x;
  const unsigned loff = (unsigned)(lo << 4);       // 16B per lane within a row
  const int spreadSrc = ((lane >> 3) & 1) << 4;    // lanes 0-7 <- lane 0, 8-15 <- lane 16

  // --- persistent loop over 128-edge block tiles, index stream pipelined ---
  int bt = blockIdx.x;
  int nidx = 0;
  if (bt < nTiles) {
    int epos = bt * 128 + w * 16 + lo; if (epos >= E) epos = E - 1;
    nidx = ei[hi * E + epos];          // lanes 0-15: row[], 16-31: col[]
  }

  for (; bt < nTiles; bt += gridDim.x) {
    const int eb  = bt * 128 + w * 16;    // this wave's 16-edge base
    const int nbt = bt + gridDim.x;

    // issue next tile's index load early (consumed at loop bottom)
    int nidx_next = 0;
    if (nbt < nTiles) {
      int np = nbt * 128 + w * 16 + lo; if (np >= E) np = E - 1;
      nidx_next = ei[hi * E + np];
    }

    // ---- gather: hoist shuffles, 32-bit voffset addressing, 16 loads in flight ----
    int nodes[16];
    #pragma unroll
    for (int ee = 0; ee < 16; ++ee)
      nodes[ee] = __shfl(nidx, ee + (hi << 4), 32);

    v4f f[16];
    #pragma unroll
    for (int ee = 0; ee < 16; ++ee) {
      const unsigned off = ((unsigned)nodes[ee] << 8) + loff;   // node*256B + lane slice
      f[ee] = *(const v4f*)(xb + off);
    }

    #pragma unroll
    for (int ee = 0; ee < 16; ++ee) {
      v4h h = { (_Float16)f[ee][0], (_Float16)f[ee][1],
                (_Float16)f[ee][2], (_Float16)f[ee][3] };
      *(v4h*)&aRow[ee * A_ROW + hi * 64 + lo * 4] = h;
    }

    // ---- GEMM1: [16x128] x [128x32] = 2 N-tiles x 4 K-steps ----
    v8f c0 = splat8(bias1a);
    v8f c1 = splat8(bias1b);
    #pragma unroll
    for (int kk = 0; kk < 4; ++kk) {
      const _Float16* ap = &aRow[lo * A_ROW + kk * 32 + hi * 16];
      const v16h a = cat8(*(const v8h*)ap, *(const v8h*)(ap + 8));
      c0 = __builtin_amdgcn_wmma_f32_16x16x32_f16(false, a, false, bw1[0][kk],
                                                  (short)0, c0, false, false);
      c1 = __builtin_amdgcn_wmma_f32_16x16x32_f16(false, a, false, bw1[1][kk],
                                                  (short)0, c1, false, false);
    }

    // ReLU + restage h1 in pi-interleaved layout: (feat lo, feat 16+lo) adjacent
    // -> v_cvt_pk_f16_f32 + single b32 store per v
    #pragma unroll
    for (int v = 0; v < 8; ++v) {
      const int m = hi * 8 + v;
      v2h p = { (_Float16)relu(c0[v]), (_Float16)relu(c1[v]) };
      *(v2h*)&h1Row[m * H1_ROW + 2 * lo] = p;
    }

    // ---- GEMM2: [16x32] x [32x16], single wmma (pi-matched B) ----
    v8f c2 = splat8(bias2);
    {
      const _Float16* ap = &h1Row[lo * H1_ROW + hi * 16];
      const v16h a = cat8(*(const v8h*)ap, *(const v8h*)(ap + 8));
      c2 = __builtin_amdgcn_wmma_f32_16x16x32_f16(false, a, false, bw2,
                                                  (short)0, c2, false, false);
    }

    // ReLU + restage h2 (features 0..15 plain); K=16..31 slots hold stale
    // finite f16 h1 data which meets zero B fragments -> contributes 0.
    #pragma unroll
    for (int v = 0; v < 8; ++v)
      h1Row[(hi * 8 + v) * H1_ROW + lo] = (_Float16)relu(c2[v]);

    // ---- GEMM3 as WMMA: scores[m] = h2[m]·W3 + b3 land in column n=0 ----
    v8f c3 = splat8(b3v);
    {
      const _Float16* ap = &h1Row[lo * H1_ROW + hi * 16];
      const v16h a = cat8(*(const v8h*)ap, *(const v8h*)(ap + 8));
      c3 = __builtin_amdgcn_wmma_f32_16x16x32_f16(false, a, false, bw3,
                                                  (short)0, c3, false, false);
    }

    // spread the 16 scores (lanes 0/16, elems 0..7) onto lanes 0..15,
    // one sigmoid chain, coalesced 64B store
    float s = 0.0f;
    #pragma unroll
    for (int v = 0; v < 8; ++v) {
      const float t = __shfl(c3[v], spreadSrc, 32);
      if ((lane & 7) == v) s = t;
    }
    const float r = sigm(sigm(s) * itemp);
    if (hi == 0) {
      const int e = eb + lo;
      if (e < E) out[e] = r;
    }

    // prefetch ALL next-tile rows: each lane owns one node index,
    // 2 prefetches cover the 256B row (2 x 128B lines)
    if (nbt < nTiles) {
      const float* pf = &x[nidx_next * 64];
      __builtin_prefetch(pf, 0, 0);
      __builtin_prefetch(pf + 32, 0, 0);
    }
    nidx = nidx_next;
  }
}

extern "C" void kernel_launch(void* const* d_in, const int* in_sizes, int n_in,
                              void* d_out, int out_size, void* d_ws, size_t ws_size,
                              hipStream_t stream) {
  const float* x    = (const float*)d_in[0];
  const int*   ei   = (const int*)  d_in[1];   // jax default x64-off -> int32
  const float* W1   = (const float*)d_in[2];
  const float* b1   = (const float*)d_in[3];
  const float* W2   = (const float*)d_in[4];
  const float* b2   = (const float*)d_in[5];
  const float* W3   = (const float*)d_in[6];
  const float* b3   = (const float*)d_in[7];
  const float* temp = (const float*)d_in[8];
  float* out = (float*)d_out;

  const int E = in_sizes[1] / 2;
  const int nTiles = (E + 127) / 128;
  const int grid = nTiles < 1280 ? nTiles : 1280;

  edge_mlp_wmma_kernel<<<grid, 256, 0, stream>>>(
      x, ei, W1, b1, W2, b2, W3, b3, temp, out, E, nTiles);
}
